// RoutedExperts_10024453669615
// MI455X (gfx1250) — matrix operand order repaired and checked by reading, not compiled, not run on hardware
//
#include <hip/hip_runtime.h>
#include <hip/hip_bf16.h>

// ---------------------------------------------------------------------------
// MoE routed experts: per-expert  h = up * silu(gate),  out = h @ w_down
//   x         : (N, D)       f32   N=16384, D=2048
//   w_up_gate : (E, 2H, D)   f32   E=16, H=1024
//   w_down    : (E, H, D)    f32
//   out       : (N, D)       f32
// Strategy: bf16 WMMA (v_wmma_f32_16x16x32_bf16) with f32 accumulation.
//   Pass1: double-buffered LDS + 2-deep register pipeline (one barrier/step)
//   Pass2: double-buffered GLOBAL_LOAD_ASYNC_TO_LDS staging (ASYNCcnt)
// ---------------------------------------------------------------------------

typedef __bf16 bf16_t;
typedef __attribute__((ext_vector_type(16))) __bf16 v16bf;
typedef __attribute__((ext_vector_type(8)))  __bf16 v8bf;
typedef __attribute__((ext_vector_type(4)))  __bf16 v4bf;
typedef __attribute__((ext_vector_type(8)))  float  v8f;
typedef __attribute__((ext_vector_type(4)))  float  v4f;
typedef int v4i __attribute__((vector_size(16)));   // matches async builtin param

#define NTOK 16384
#define DDIM 2048
#define HDIM 1024
#define NEXP 16
#define TPE  1024   // tokens per expert

#define BM   128    // block tile: output-feature dim (WMMA M)
#define BN   128    // block tile: token dim (WMMA N)
#define BK   32     // K step (WMMA K for bf16)
#define LDK  40     // padded LDS row length in bf16 elems (80B stride: conflict-free)

#define AS1 __attribute__((address_space(1)))
#define AS3 __attribute__((address_space(3)))

#if defined(__HIP_DEVICE_COMPILE__) && \
    __has_builtin(__builtin_amdgcn_global_load_async_to_lds_b128) && \
    __has_builtin(__builtin_amdgcn_s_wait_asynccnt)
#define HAVE_ASYNC_LDS 1
#else
#define HAVE_ASYNC_LDS 0
#endif

// --- fragment loads from LDS -----------------------------------------------
// A frag (16x32 bf16, ISA 7.12.2): lane<16 holds row M=lane, K=0..7 & 16..23;
// lane>=16 holds K=8..15 & 24..31.
__device__ inline v16bf frag_a(const bf16_t* __restrict__ s, int row, int lane) {
    const bf16_t* p = s + row * LDK;
    int kh = (lane >> 4) << 3;          // 0 or 8
    v8bf lo = *(const v8bf*)(p + kh);
    v8bf hi = *(const v8bf*)(p + 16 + kh);
    return __builtin_shufflevector(lo, hi, 0,1,2,3,4,5,6,7,8,9,10,11,12,13,14,15);
}
// B frag (32x16 bf16): lane<16 holds col N=lane, K=0..7 & 16..23; lane>=16 K=8..15 & 24..31.
__device__ inline v16bf frag_b(const bf16_t* __restrict__ s, int row, int lane) {
    const bf16_t* p = s + row * LDK + ((lane >> 4) << 3);
    v8bf lo = *(const v8bf*)(p);
    v8bf hi = *(const v8bf*)(p + 16);
    return __builtin_shufflevector(lo, hi, 0,1,2,3,4,5,6,7,8,9,10,11,12,13,14,15);
}

__device__ inline void cvt_store(v4f v, bf16_t* p) {
    v4bf b;
    b[0] = (bf16_t)v[0]; b[1] = (bf16_t)v[1];
    b[2] = (bf16_t)v[2]; b[3] = (bf16_t)v[3];
    *(v4bf*)p = b;
}

#define WMMA_BF16(a, b, c) \
    __builtin_amdgcn_wmma_f32_16x16x32_bf16(false, (a), false, (b), (short)0, (c), false, false)

// ---------------------------------------------------------------------------
// Prep: wdT[e][d][h] = (bf16) w_down[e][h][d]
// ---------------------------------------------------------------------------
__global__ __launch_bounds__(256)
void transpose_wdown(const float* __restrict__ wd, bf16_t* __restrict__ wdT) {
    __shared__ float tile[32][33];
    int e  = blockIdx.z;
    int d0 = blockIdx.x * 32;
    int h0 = blockIdx.y * 32;
    int tx = threadIdx.x & 31, ty = threadIdx.x >> 5;   // ty in 0..7
    const float* src = wd + (size_t)e * HDIM * DDIM;
    #pragma unroll
    for (int i = 0; i < 4; ++i) {
        int h = h0 + ty * 4 + i;
        tile[ty * 4 + i][tx] = src[(size_t)h * DDIM + d0 + tx];
    }
    __syncthreads();
    bf16_t* dst = wdT + (size_t)e * DDIM * HDIM;
    #pragma unroll
    for (int i = 0; i < 4; ++i) {
        int d = d0 + ty * 4 + i;
        dst[(size_t)d * HDIM + h0 + tx] = (bf16_t)tile[tx][ty * 4 + i];
    }
}

// ---------------------------------------------------------------------------
// Pass 1: h[e][t][hc] = up * silu(gate)   (bf16 out)
//   WMMA:  M = h-column (weight rows), N = token, K = D
//   2-deep pipeline: global loads for tile k+2 and LDS stores for tile k+1
//   are in flight while the WMMA burst for tile k executes; one barrier/step.
// ---------------------------------------------------------------------------
struct Pref1 { v4f u[2]; v4f g[2]; v4f b[2]; };

__device__ inline Pref1 load1(const float* __restrict__ wu, const float* __restrict__ wg,
                              const float* __restrict__ xb, int k, int tid) {
    Pref1 p;
    #pragma unroll
    for (int i = 0; i < 2; ++i) {
        int idx = tid + i * 512;
        int row = idx >> 3;
        int c4  = (idx & 7) << 2;
        p.u[i] = *(const v4f*)(wu + (size_t)row * DDIM + k + c4);
        p.g[i] = *(const v4f*)(wg + (size_t)row * DDIM + k + c4);
        p.b[i] = *(const v4f*)(xb + (size_t)row * DDIM + k + c4);
    }
    return p;
}
__device__ inline void store1(const Pref1& p, bf16_t* sAu, bf16_t* sAg, bf16_t* sB, int tid) {
    #pragma unroll
    for (int i = 0; i < 2; ++i) {
        int idx = tid + i * 512;
        int row = idx >> 3;
        int c4  = (idx & 7) << 2;
        cvt_store(p.u[i], sAu + row * LDK + c4);
        cvt_store(p.g[i], sAg + row * LDK + c4);
        cvt_store(p.b[i], sB  + row * LDK + c4);
    }
}

__global__ __launch_bounds__(512)
void moe_up_gate_silu(const float* __restrict__ x, const float* __restrict__ wug,
                      bf16_t* __restrict__ hbuf) {
    __shared__ bf16_t sAu[2][BM * LDK];
    __shared__ bf16_t sAg[2][BM * LDK];
    __shared__ bf16_t sB [2][BN * LDK];

    int e   = blockIdx.z;
    int m0  = blockIdx.x * BM;            // h-column base
    int n0  = blockIdx.y * BN;            // token base within expert
    int tid = threadIdx.x;
    int wave = tid >> 5, lane = tid & 31;
    int wm = wave & 3, wn = wave >> 2;    // 4x4 wave grid: 32x32 per wave

    const float* wu = wug + (size_t)e * (2 * HDIM) * DDIM + (size_t)m0 * DDIM;
    const float* wg = wu + (size_t)HDIM * DDIM;
    const float* xb = x + ((size_t)e * TPE + n0) * DDIM;

    v8f acc_u[2][2] = {};
    v8f acc_g[2][2] = {};

    // Pipeline prologue: tile 0 staged to buf 0; tile 1 loads in flight.
    Pref1 pnext = load1(wu, wg, xb, 0, tid);
    store1(pnext, sAu[0], sAg[0], sB[0], tid);
    pnext = load1(wu, wg, xb, BK, tid);
    int cur = 0;

    for (int k = 0; k < DDIM; k += BK) {
        __syncthreads();  // buf[cur] stores visible; buf[1-cur] reads drained
        if (k + BK < DDIM) {
            store1(pnext, sAu[1 - cur], sAg[1 - cur], sB[1 - cur], tid);
            if (k + 2 * BK < DDIM)
                pnext = load1(wu, wg, xb, k + 2 * BK, tid);
        }

        int arow = wm * 32 + (lane & 15);
        int brow = wn * 32 + (lane & 15);
        #pragma unroll
        for (int nt = 0; nt < 2; ++nt) {
            v16bf fb = frag_b(sB[cur], brow + nt * 16, lane);
            #pragma unroll
            for (int mt = 0; mt < 2; ++mt) {
                v16bf au = frag_a(sAu[cur], arow + mt * 16, lane);
                v16bf ag = frag_a(sAg[cur], arow + mt * 16, lane);
                acc_u[mt][nt] = WMMA_BF16(au, fb, acc_u[mt][nt]);
                acc_g[mt][nt] = WMMA_BF16(ag, fb, acc_g[mt][nt]);
            }
        }
        cur ^= 1;
    }

    // Epilogue: lane holds fixed token (N=lane&15), 8 consecutive h-columns
    // (M = r + 8*(lane>>4)) -> one packed 16B bf16 store per frag.
    #pragma unroll
    for (int nt = 0; nt < 2; ++nt) {
        int tok = n0 + wn * 32 + nt * 16 + (lane & 15);
        #pragma unroll
        for (int mt = 0; mt < 2; ++mt) {
            int hc = m0 + wm * 32 + mt * 16 + ((lane >> 4) << 3);
            v8bf hv;
            #pragma unroll
            for (int r = 0; r < 8; ++r) {
                float g = acc_g[mt][nt][r];
                float u = acc_u[mt][nt][r];
                float sig = __builtin_amdgcn_rcpf(1.0f + __expf(-g)); // v_rcp
                hv[r] = (bf16_t)(u * g * sig);                        // u*silu(g)
            }
            *(v8bf*)(hbuf + ((size_t)e * TPE + tok) * HDIM + hc) = hv;
        }
    }
}

// ---------------------------------------------------------------------------
// Pass 2: out[e][t][d] = h[e][t][:] . wdT[e][d][:]
//   WMMA:  M = d (wdT rows), N = token, K = H
//   Double-buffered LDS staging via GLOBAL_LOAD_ASYNC_TO_LDS (ASYNCcnt).
// ---------------------------------------------------------------------------
__device__ inline void stage2(const bf16_t* __restrict__ g, int g_ld,
                              bf16_t* __restrict__ s, int tid) {
    // 128 rows x 32 bf16; one 16B chunk per thread (512 threads)
    int row = tid >> 2;
    int c8  = (tid & 3) << 3;
    const bf16_t* gp = g + (size_t)row * g_ld + c8;
    bf16_t*       lp = s + row * LDK + c8;
#if HAVE_ASYNC_LDS
    __builtin_amdgcn_global_load_async_to_lds_b128(
        (AS1 v4i*)(gp), (AS3 v4i*)(lp), 0, 0);
#else
    *(v8bf*)lp = *(const v8bf*)gp;
#endif
}
__device__ inline void wait_async_lds() {
#if HAVE_ASYNC_LDS
    __builtin_amdgcn_s_wait_asynccnt(0);
#endif
}

__global__ __launch_bounds__(512)
void moe_down(const bf16_t* __restrict__ wdT, const bf16_t* __restrict__ hbuf,
              float* __restrict__ out) {
    __shared__ bf16_t sA[2][BM * LDK];
    __shared__ bf16_t sB[2][BN * LDK];

    int e   = blockIdx.z;
    int m0  = blockIdx.x * BM;            // d base
    int n0  = blockIdx.y * BN;            // token base within expert
    int tid = threadIdx.x;
    int wave = tid >> 5, lane = tid & 31;
    int wm = wave & 3, wn = wave >> 2;

    const bf16_t* A = wdT  + ((size_t)e * DDIM + m0) * HDIM;
    const bf16_t* B = hbuf + ((size_t)e * TPE  + n0) * HDIM;

    v8f acc[2][2] = {};

    stage2(A, HDIM, sA[0], tid);          // prologue: stage K-step 0 -> buf 0
    stage2(B, HDIM, sB[0], tid);
    int cur = 0;

    for (int k = 0; k < HDIM; k += BK) {
        wait_async_lds();                 // buf[cur] staged (this wave's part)
        __syncthreads();                  // all waves staged + prev reads done
        if (k + BK < HDIM) {              // prefetch next tile into buf[1-cur]
            stage2(A + k + BK, HDIM, sA[1 - cur], tid);
            stage2(B + k + BK, HDIM, sB[1 - cur], tid);
        }

        int arow = wm * 32 + (lane & 15);
        int brow = wn * 32 + (lane & 15);
        #pragma unroll
        for (int nt = 0; nt < 2; ++nt) {
            v16bf fb = frag_b(sB[cur], brow + nt * 16, lane);
            #pragma unroll
            for (int mt = 0; mt < 2; ++mt) {
                v16bf fa = frag_a(sA[cur], arow + mt * 16, lane);
                acc[mt][nt] = WMMA_BF16(fa, fb, acc[mt][nt]);
            }
        }
        cur ^= 1;
    }

    // Lane holds fixed token, 8 consecutive d values -> two b128 f32 stores.
    #pragma unroll
    for (int nt = 0; nt < 2; ++nt) {
        int tok = n0 + wn * 32 + nt * 16 + (lane & 15);
        #pragma unroll
        for (int mt = 0; mt < 2; ++mt) {
            int d = m0 + wm * 32 + mt * 16 + ((lane >> 4) << 3);
            float* p = out + ((size_t)e * TPE + tok) * DDIM + d;
            v8f a = acc[mt][nt];
            v4f lo = {a[0], a[1], a[2], a[3]};
            v4f hi = {a[4], a[5], a[6], a[7]};
            *(v4f*)(p)     = lo;
            *(v4f*)(p + 4) = hi;
        }
    }
}

// ---------------------------------------------------------------------------
extern "C" void kernel_launch(void* const* d_in, const int* in_sizes, int n_in,
                              void* d_out, int out_size, void* d_ws, size_t ws_size,
                              hipStream_t stream) {
    const float* x   = (const float*)d_in[0];   // (N, D)
    const float* wug = (const float*)d_in[1];   // (E, 2H, D)
    const float* wd  = (const float*)d_in[2];   // (E, H, D)
    // d_in[3] = batch_size_per_expert (constant N/E; unused by reference math)
    float* out = (float*)d_out;

    // workspace layout: [ wdT bf16 : E*D*H ][ hbuf bf16 : E*T*H ]
    bf16_t* wdT  = (bf16_t*)d_ws;
    bf16_t* hbuf = (bf16_t*)((char*)d_ws + (size_t)NEXP * DDIM * HDIM * sizeof(bf16_t));

    dim3 gt(DDIM / 32, HDIM / 32, NEXP);
    transpose_wdown<<<gt, 256, 0, stream>>>(wd, wdT);

    dim3 g1(HDIM / BM, TPE / BN, NEXP);
    moe_up_gate_silu<<<g1, 512, 0, stream>>>(x, wug, hbuf);

    dim3 g2(DDIM / BM, TPE / BN, NEXP);
    moe_down<<<g2, 512, 0, stream>>>(wdT, hbuf, out);
}
